// ExpertBank_31181462569498
// MI455X (gfx1250) — matrix operand order repaired and checked by reading, not compiled, not run on hardware
//
#include <hip/hip_runtime.h>
#include <hip/hip_bf16.h>

typedef __attribute__((ext_vector_type(16))) __bf16 v16bf;
typedef __attribute__((ext_vector_type(8)))  __bf16 v8bf;
typedef __attribute__((ext_vector_type(4)))  __bf16 v4bf;
typedef __attribute__((ext_vector_type(8)))  float  v8f;

#define BM 128
#define BN 128
#define BK 32
#define LDA 40   // padded row stride (elems) for A tile [BM][LDA]  (80B)
#define LDB 40   // padded col stride (elems) for B tiles [BN][LDB] (N-major, K contiguous)
#define NBUF 2

union Frag { v16bf v; v8bf h[2]; };

static __device__ __forceinline__ __bf16 tobf(float f) { return (__bf16)f; }

// 256 threads = 8 waves; waves arranged 4 (M) x 2 (N); each wave = 32x64 = 2x4 WMMA frags.
// Both experts (K=2) fused into one D-loop sharing the A tile; LDS double-buffered.
__global__ __launch_bounds__(256)
void moe_expert_wmma_kernel(const float* __restrict__ x,
                            const float* __restrict__ w,
                            const float* __restrict__ W_bank,
                            const float* __restrict__ b_bank,
                            const int*   __restrict__ idx,
                            float* __restrict__ y,
                            int B, int T, int D)
{
    __shared__ __bf16 ldsA[NBUF][BM * LDA];
    __shared__ __bf16 ldsB[NBUF][2][BN * LDB];

    const int tid    = threadIdx.x;
    const int lane   = tid & 31;
    const int wid    = tid >> 5;
    const int wave_m = wid & 3;          // 0..3 -> M offset wave_m*32
    const int wave_n = wid >> 2;         // 0..1 -> N offset wave_n*64
    const int l16    = lane & 15;
    const int lhalf  = lane >> 4;        // 0/1

    const int tilesN = D / BN;
    const int tilesM = T / BM;
    int bid = blockIdx.x;
    const int tn = bid % tilesN;  bid /= tilesN;
    const int tm = bid % tilesM;  bid /= tilesM;
    const int b  = bid;

    const int m0 = tm * BM;
    const int n0 = tn * BN;

    const int   e0 = idx[b * 2 + 0];
    const int   e1 = idx[b * 2 + 1];
    const float w0 = w[b * 2 + 0];
    const float w1 = w[b * 2 + 1];
    const float* __restrict__ W0 = W_bank + (size_t)e0 * D * D;
    const float* __restrict__ W1 = W_bank + (size_t)e1 * D * D;

    // staging coordinates
    const int arow = tid >> 1;           // 0..127
    const int aseg = (tid & 1) * 16;     // 0 / 16
    const int bo4  = (tid & 31) * 4;     // 0..124 (N, coalesced)
    const int bi0  = (tid >> 5) * 4;     // 0..28  (K rows, 4 per thread)

    v8f acc0[2][4] = {};
    v8f acc1[2][4] = {};

    float4 ra[4], rb0[4], rb1[4];

    auto gload = [&](int kk) {
        const float* gp = x + ((size_t)b * T + m0 + arow) * D + kk + aseg;
        ra[0] = ((const float4*)gp)[0];
        ra[1] = ((const float4*)gp)[1];
        ra[2] = ((const float4*)gp)[2];
        ra[3] = ((const float4*)gp)[3];
        #pragma unroll
        for (int q = 0; q < 4; ++q) {
            rb0[q] = *(const float4*)(W0 + (size_t)(kk + bi0 + q) * D + n0 + bo4);
            rb1[q] = *(const float4*)(W1 + (size_t)(kk + bi0 + q) * D + n0 + bo4);
        }
    };

    auto sstore = [&](int buf) {
        // A tile: row-major bf16, two ds_store_b128
        v8bf p0 = { tobf(ra[0].x), tobf(ra[0].y), tobf(ra[0].z), tobf(ra[0].w),
                    tobf(ra[1].x), tobf(ra[1].y), tobf(ra[1].z), tobf(ra[1].w) };
        v8bf p1 = { tobf(ra[2].x), tobf(ra[2].y), tobf(ra[2].z), tobf(ra[2].w),
                    tobf(ra[3].x), tobf(ra[3].y), tobf(ra[3].z), tobf(ra[3].w) };
        v8bf* dp = (v8bf*)&ldsA[buf][arow * LDA + aseg];
        dp[0] = p0;
        dp[1] = p1;
        // B tiles: transpose to N-major, 4 K-values packed -> ds_store_b64 per column
        const float* f0 = (const float*)rb0;   // [row q][col c] = f0[q*4+c]
        const float* f1 = (const float*)rb1;
        #pragma unroll
        for (int q = 0; q < 4; ++q) {          // q = column within float4 (N)
            v4bf c0 = { tobf(f0[0*4+q]), tobf(f0[1*4+q]), tobf(f0[2*4+q]), tobf(f0[3*4+q]) };
            v4bf c1 = { tobf(f1[0*4+q]), tobf(f1[1*4+q]), tobf(f1[2*4+q]), tobf(f1[3*4+q]) };
            *(v4bf*)&ldsB[buf][0][(bo4 + q) * LDB + bi0] = c0;
            *(v4bf*)&ldsB[buf][1][(bo4 + q) * LDB + bi0] = c1;
        }
    };

    auto compute = [&](int buf) {
        // A fragments (16x32 bf16): lane<16 -> K0-7 & K16-23; lane>=16 -> K8-15 & K24-31
        Frag af[2];
        #pragma unroll
        for (int i = 0; i < 2; ++i) {
            const int row = wave_m * 32 + i * 16 + l16;
            af[i].h[0] = *(const v8bf*)&ldsA[buf][row * LDA + lhalf * 8];
            af[i].h[1] = *(const v8bf*)&ldsA[buf][row * LDA + 16 + lhalf * 8];
        }
        // Expert 0
        {
            Frag bfr[4];
            #pragma unroll
            for (int j = 0; j < 4; ++j) {
                const int col = wave_n * 64 + j * 16 + l16;
                const v8bf* p = (const v8bf*)&ldsB[buf][0][col * LDB + lhalf * 8];
                bfr[j].h[0] = p[0];
                bfr[j].h[1] = p[2];
            }
            #pragma unroll
            for (int i = 0; i < 2; ++i)
                #pragma unroll
                for (int j = 0; j < 4; ++j)
                    acc0[i][j] = __builtin_amdgcn_wmma_f32_16x16x32_bf16(
                        false, af[i].v, false, bfr[j].v,
                        (short)0, acc0[i][j], false, false);
        }
        // Expert 1
        {
            Frag bfr[4];
            #pragma unroll
            for (int j = 0; j < 4; ++j) {
                const int col = wave_n * 64 + j * 16 + l16;
                const v8bf* p = (const v8bf*)&ldsB[buf][1][col * LDB + lhalf * 8];
                bfr[j].h[0] = p[0];
                bfr[j].h[1] = p[2];
            }
            #pragma unroll
            for (int i = 0; i < 2; ++i)
                #pragma unroll
                for (int j = 0; j < 4; ++j)
                    acc1[i][j] = __builtin_amdgcn_wmma_f32_16x16x32_bf16(
                        false, af[i].v, false, bfr[j].v,
                        (short)0, acc1[i][j], false, false);
        }
    };

    // ---- software pipeline over D: double-buffered, one barrier per step
    const int S = D / BK;
    gload(0);
    sstore(0);
    __syncthreads();
    for (int s = 0; s < S; ++s) {
        if (s + 1 < S) gload((s + 1) * BK);   // global loads in flight during WMMAs
        compute(s & 1);
        if (s + 1 < S) sstore((s + 1) & 1);   // fill the other buffer
        __syncthreads();
    }

    // ---- epilogue: bias + relu per expert, weighted combine, saddr-form stores
    float* __restrict__ yb = y + ((size_t)b * T + m0) * D + n0;
    #pragma unroll
    for (int j = 0; j < 4; ++j) {
        const int   cl    = wave_n * 64 + j * 16 + l16;      // col within tile
        const float bias0 = b_bank[(size_t)e0 * D + n0 + cl];
        const float bias1 = b_bank[(size_t)e1 * D + n0 + cl];
        #pragma unroll
        for (int i = 0; i < 2; ++i) {
            #pragma unroll
            for (int r = 0; r < 8; ++r) {
                const int row = wave_m * 32 + i * 16 + lhalf * 8 + r;
                float v0 = acc0[i][j][r] + bias0;  v0 = v0 > 0.f ? v0 : 0.f;
                float v1 = acc1[i][j][r] + bias1;  v1 = v1 > 0.f ? v1 : 0.f;
                yb[(unsigned)(row * D + cl)] = w0 * v0 + w1 * v1;
            }
        }
    }
}

extern "C" void kernel_launch(void* const* d_in, const int* in_sizes, int n_in,
                              void* d_out, int out_size, void* d_ws, size_t ws_size,
                              hipStream_t stream) {
    const float* x      = (const float*)d_in[0];   // (B,T,D)
    const float* w      = (const float*)d_in[1];   // (B,K=2)
    const float* W_bank = (const float*)d_in[2];   // (E,D,D)
    const float* b_bank = (const float*)d_in[3];   // (E,D)
    const int*   idx    = (const int*)d_in[4];     // (B,K=2)
    float*       y      = (float*)d_out;           // (B,T,D)

    const int K = 2;
    const int B = in_sizes[1] / K;
    const long long sW = in_sizes[2];              // E*D*D
    const long long sb = in_sizes[3];              // E*D
    const int D = (int)(sW / sb);
    const int T = (int)((long long)in_sizes[0] / ((long long)B * D));

    dim3 block(256);
    dim3 grid((unsigned)(B * (T / BM) * (D / BN)));
    moe_expert_wmma_kernel<<<grid, block, 0, stream>>>(x, w, W_bank, b_bank, idx, y,
                                                       B, T, D);
}
